// SelfAttentionEncoder_62886911148687
// MI455X (gfx1250) — compile-verified, matrix operand order checked by reading, not run on hardware
//
#include <hip/hip_runtime.h>
#include <hip/hip_bf16.h>
#include <math.h>

// ---------------------------------------------------------------------------
// Transformer-XL encoder forward on gfx1250 (MI455X), compile-only target.
// All matmuls use V_WMMA_F32_16X16X32_F16; all global operand streams are
// contiguous b128 loads (weights pre-transposed to f16 [N][K]).
// ---------------------------------------------------------------------------

#define B_  2
#define L_  1024
#define IN_ 512
#define HID_ 1024
#define H_  16
#define DH_ 64
#define NL_ 2

typedef __attribute__((ext_vector_type(16))) _Float16 v16h;
typedef __attribute__((ext_vector_type(8)))  _Float16 h8;
typedef __attribute__((ext_vector_type(8)))  float    f8;
typedef __attribute__((ext_vector_type(8)))  float    v8f;

__device__ __forceinline__ float gelu_f(float x) {
    return 0.5f * x * (1.0f + erff(x * 0.70710678118654752f));
}

// Per-lane fragment: element e covers K = kbase + {0..7} (e<8) / {16..23} (e>=8),
// kbase = 0 for lanes 0-15, 8 for lanes 16-31. Both runs contiguous -> b128s.
__device__ __forceinline__ v16h loadA_f32(const float* p /* at k0+kbase */) {
    const f8* q = (const f8*)p;
    f8 lo = q[0];          // +0..7
    f8 hi = q[2];          // +16..23
    v16h r;
#pragma unroll
    for (int e = 0; e < 8; ++e) { r[e] = (_Float16)lo[e]; r[e + 8] = (_Float16)hi[e]; }
    return r;
}
__device__ __forceinline__ v16h loadB_f16(const _Float16* p /* at k0+kbase */) {
    const h8* q = (const h8*)p;
    h8 lo = q[0];
    h8 hi = q[2];
    return __builtin_shufflevector(lo, hi, 0,1,2,3,4,5,6,7,8,9,10,11,12,13,14,15);
}

// ---------------------------------------------------------------------------
// Weight repack: dst[N][K] f16.
//   mode 0: src is f32 row-major [K][N]
//   mode 1: src is head-packed f32 [H][K][64], n = h*64+d
// ---------------------------------------------------------------------------
__global__ void repack_w(const float* __restrict__ src, _Float16* __restrict__ dst,
                         int K, int N, int mode)
{
    size_t idx = (size_t)blockIdx.x * 256 + threadIdx.x;   // N*K threads
    int k = (int)(idx % K);
    int n = (int)(idx / K);
    float v = (mode == 0) ? src[(size_t)k * N + n]
                          : src[((size_t)(n >> 6) * K + k) * 64 + (n & 63)];
    dst[idx] = (_Float16)v;
}

// ---------------------------------------------------------------------------
// NT GEMM: C[M,N] = act( sum_k A[m,k] * BT[n,k] + bias[n] )
// One wave computes a 16(M) x 64(N) strip: 4 accumulators share one A frag.
// blockDim = 128 (4 waves); grid = (M/16)*(N/64)/4.
// ---------------------------------------------------------------------------
template <int ACT>
__global__ void gemm_nt(const float* __restrict__ A, int lda,
                        const _Float16* __restrict__ BT, int ldb,
                        const float* __restrict__ bias,
                        float* __restrict__ C, int ldc,
                        int M, int N, int K)
{
    int tiles_n = N >> 6;
    int wid = (int)((blockIdx.x * blockDim.x + threadIdx.x) >> 5);
    int tn = wid % tiles_n;
    int tm = wid / tiles_n;
    int lane  = threadIdx.x & 31;
    int r16   = lane & 15;
    int kbase = (lane & 16) ? 8 : 0;
    int mrow  = (lane & 16) ? 8 : 0;

    const float*    arow = A  + (size_t)(tm * 16 + r16) * lda + kbase;
    const _Float16* brow = BT + (size_t)(tn * 64 + r16) * ldb + kbase;

    v8f a0 = {}, a1 = {}, a2 = {}, a3 = {};
    for (int k0 = 0; k0 < K; k0 += 32) {
        v16h af = loadA_f32(arow + k0);
        v16h b0 = loadB_f16(brow + k0);
        v16h b1 = loadB_f16(brow + (size_t)16 * ldb + k0);
        v16h b2 = loadB_f16(brow + (size_t)32 * ldb + k0);
        v16h b3 = loadB_f16(brow + (size_t)48 * ldb + k0);
        a0 = __builtin_amdgcn_wmma_f32_16x16x32_f16(false, af, false, b0, (short)0, a0, false, false);
        a1 = __builtin_amdgcn_wmma_f32_16x16x32_f16(false, af, false, b1, (short)0, a1, false, false);
        a2 = __builtin_amdgcn_wmma_f32_16x16x32_f16(false, af, false, b2, (short)0, a2, false, false);
        a3 = __builtin_amdgcn_wmma_f32_16x16x32_f16(false, af, false, b3, (short)0, a3, false, false);
    }

    v8f accs[4] = {a0, a1, a2, a3};
#pragma unroll
    for (int nt = 0; nt < 4; ++nt) {
        int nn = tn * 64 + nt * 16 + r16;
        float bs = bias[nn];
#pragma unroll
        for (int e = 0; e < 8; ++e) {
            int mm = tm * 16 + e + mrow;
            float v = accs[nt][e] + bs;
            if (ACT) v = gelu_f(v);
            C[(size_t)mm * ldc + nn] = v;
        }
    }
}

// ---------------------------------------------------------------------------
// vT[b][h][d][j] = v[b][j][h*DH+d]
// ---------------------------------------------------------------------------
__global__ void transpose_v_kernel(const float* __restrict__ v, float* __restrict__ vT)
{
    int idx = blockIdx.x * blockDim.x + threadIdx.x;
    int j = idx & (L_ - 1);
    int d = (idx >> 10) & (DH_ - 1);
    int h = (idx >> 16) & (H_ - 1);
    int b = idx >> 20;
    vT[idx] = v[((size_t)(b * L_ + j)) * HID_ + h * DH_ + d];
}

// ---------------------------------------------------------------------------
// Fused attention: one wave per (b, h, 16-query-row block).
//   sh[16][1024] fp32 score strip (64KB), shp[16][1024] f16 probs (32KB),
//   qsh[64] staging for the rel-shift patch row (no register array -> no spill).
//     AC = (q+cbias).K^T via WMMA; BD = rel_shift((q+pbias).P^T) via WMMA +
//     exact shift scatter; softmax(.*1/8) -> f16 probs; O = P.V via WMMA.
// ---------------------------------------------------------------------------
__global__ void attn_kernel(const float* __restrict__ q,
                            const float* __restrict__ kbuf,
                            const float* __restrict__ vT,
                            const float* __restrict__ Pm,
                            const float* __restrict__ cb,
                            const float* __restrict__ pb,
                            float* __restrict__ out)
{
    __shared__ __align__(32) float     sh[16 * L_];
    __shared__ __align__(16) _Float16  shp[16 * L_];
    __shared__ float qsh[DH_];

    int wg = blockIdx.x;
    int i0 = (wg & 63) * 16;
    int h  = (wg >> 6) & (H_ - 1);
    int b  = wg >> 10;

    int lane  = threadIdx.x & 31;
    int r16   = lane & 15;
    int kbase = (lane & 16) ? 8 : 0;
    int mrow  = (lane & 16) ? 8 : 0;

    const float* cbh  = cb + h * DH_;
    const float* pbh  = pb + h * DH_;
    const float* qrow = q + ((size_t)(b * L_ + i0 + r16)) * HID_ + h * DH_;

    // Q fragments (content/position biased), DH=64 -> two K=32 fragments
    v16h aq0, aq1, ap0, ap1;
#pragma unroll
    for (int e = 0; e < 16; ++e) {
        int kk = kbase + (e < 8 ? e : e + 8);
        float q0 = qrow[kk], q1 = qrow[kk + 32];
        aq0[e] = (_Float16)(q0 + cbh[kk]);
        aq1[e] = (_Float16)(q1 + cbh[kk + 32]);
        ap0[e] = (_Float16)(q0 + pbh[kk]);
        ap1[e] = (_Float16)(q1 + pbh[kk + 32]);
    }

    // ---- content scores AC ----
    for (int jb = 0; jb < 64; ++jb) {
        const float* krow = kbuf + ((size_t)(b * L_ + jb * 16 + r16)) * HID_ + h * DH_ + kbase;
        v16h b0 = loadA_f32(krow);
        v16h b1 = loadA_f32(krow + 32);
        v8f c = {};
        c = __builtin_amdgcn_wmma_f32_16x16x32_f16(false, aq0, false, b0, (short)0, c, false, false);
        c = __builtin_amdgcn_wmma_f32_16x16x32_f16(false, aq1, false, b1, (short)0, c, false, false);
#pragma unroll
        for (int e = 0; e < 8; ++e)
            sh[(e + mrow) * L_ + jb * 16 + r16] = c[e];
    }
    __syncthreads();

    // ---- position scores: BDraw + exact rel-shift scatter ----
    // element (row rr, pos p): p <= L-rr-2 -> (rr-1, p+rr+1); else (rr, p+rr+1-L)
    for (int pbk = 0; pbk < 64; ++pbk) {
        const float* prow = Pm + ((size_t)(pbk * 16 + r16)) * HID_ + h * DH_ + kbase;
        v16h b0 = loadA_f32(prow);
        v16h b1 = loadA_f32(prow + 32);
        v8f c = {};
        c = __builtin_amdgcn_wmma_f32_16x16x32_f16(false, ap0, false, b0, (short)0, c, false, false);
        c = __builtin_amdgcn_wmma_f32_16x16x32_f16(false, ap1, false, b1, (short)0, c, false, false);
#pragma unroll
        for (int e = 0; e < 8; ++e) {
            int M  = e + mrow;
            int rr = i0 + M;
            int p  = pbk * 16 + r16;
            if (p <= L_ - rr - 2) {
                if (M > 0) sh[(M - 1) * L_ + p + rr + 1] += c[e];
            } else {
                sh[M * L_ + (p + rr + 1 - L_)] += c[e];
            }
        }
    }
    // missing source row i0+16 feeds local row 15's cols j >= i0+17
    {
        int rr2 = i0 + 16;
        if (rr2 < L_) {
            const float* q2 = q + ((size_t)(b * L_ + rr2)) * HID_ + h * DH_;
            qsh[lane]      = q2[lane]      + pbh[lane];
            qsh[lane + 32] = q2[lane + 32] + pbh[lane + 32];
            __syncthreads();
            int pmax = L_ - rr2 - 2;
            for (int p = lane; p <= pmax; p += 32) {
                const float* prow = Pm + (size_t)p * HID_ + h * DH_;
                float s = 0.0f;
#pragma unroll 8
                for (int kk = 0; kk < DH_; ++kk) s += qsh[kk] * prow[kk];
                sh[15 * L_ + p + rr2 + 1] += s;
            }
        }
    }
    __syncthreads();

    // ---- softmax per row (2 lanes per row); emit f16 probs to shp ----
    {
        const float scale = 0.125f;   // 1/sqrt(DH); commutes with max
        int r  = lane >> 1;
        int c0 = (lane & 1) * 512;
        float mx = -1e30f;
        for (int c = 0; c < 512; ++c) mx = fmaxf(mx, sh[r * L_ + c0 + c]);
        mx = fmaxf(mx, __shfl_xor(mx, 1, 32));
        float sum = 0.0f;
        for (int c = 0; c < 512; ++c) {
            float ev = __expf((sh[r * L_ + c0 + c] - mx) * scale);
            sum += ev;
            sh[r * L_ + c0 + c] = ev;
        }
        sum += __shfl_xor(sum, 1, 32);
        float inv = 1.0f / sum;
        for (int c = 0; c < 512; ++c) {
            int idx = r * L_ + c0 + c;
            shp[idx] = (_Float16)(sh[idx] * inv);
        }
    }
    __syncthreads();

    // ---- O = P.V (f16 probs from LDS, V^T rows contiguous) ----
    for (int nt = 0; nt < 4; ++nt) {
        const float* vrow = vT + ((size_t)(b * H_ + h) * DH_ + nt * 16 + r16) * L_ + kbase;
        v8f c = {};
        for (int ks = 0; ks < L_ / 32; ++ks) {
            v16h a  = loadB_f16(&shp[r16 * L_ + ks * 32 + kbase]);
            v16h bb = loadA_f32(vrow + ks * 32);
            c = __builtin_amdgcn_wmma_f32_16x16x32_f16(false, a, false, bb, (short)0, c, false, false);
        }
#pragma unroll
        for (int e = 0; e < 8; ++e)
            out[((size_t)(b * L_ + i0 + e + mrow)) * HID_ + h * DH_ + nt * 16 + r16] = c[e];
    }
}

// ---------------------------------------------------------------------------
// out[row] = LN(A[row] + B[row]), BertLayerNorm (eps inside sqrt)
// ---------------------------------------------------------------------------
__global__ void resid_ln_kernel(const float* __restrict__ A, const float* __restrict__ Bv,
                                const float* __restrict__ w, const float* __restrict__ bias,
                                float* __restrict__ out)
{
    const int N = HID_;
    __shared__ float red[256];
    int row = blockIdx.x, tid = threadIdx.x;
    const float* pa = A  + (size_t)row * N;
    const float* pb = Bv + (size_t)row * N;

    float s = 0.0f;
    for (int c = tid; c < N; c += 256) s += pa[c] + pb[c];
    red[tid] = s; __syncthreads();
    for (int off = 128; off > 0; off >>= 1) { if (tid < off) red[tid] += red[tid + off]; __syncthreads(); }
    float mean = red[0] * (1.0f / N);
    __syncthreads();

    float s2 = 0.0f;
    for (int c = tid; c < N; c += 256) { float d = pa[c] + pb[c] - mean; s2 += d * d; }
    red[tid] = s2; __syncthreads();
    for (int off = 128; off > 0; off >>= 1) { if (tid < off) red[tid] += red[tid + off]; __syncthreads(); }
    float inv = 1.0f / sqrtf(red[0] * (1.0f / N) + 1e-12f);

    for (int c = tid; c < N; c += 256) {
        float v = pa[c] + pb[c];
        out[(size_t)row * N + c] = w[c] * (v - mean) * inv + bias[c];
    }
}

// ---------------------------------------------------------------------------
extern "C" void kernel_launch(void* const* d_in, const int* in_sizes, int n_in,
                              void* d_out, int out_size, void* d_ws, size_t ws_size,
                              hipStream_t stream)
{
    const float* x       = (const float*)d_in[0];
    const float* pos_raw = (const float*)d_in[1];
    const float* Wh1     = (const float*)d_in[2];
    const float* bh1     = (const float*)d_in[3];
    const float* Wh2     = (const float*)d_in[4];
    const float* bh2     = (const float*)d_in[5];
    const float* Wq      = (const float*)d_in[6];
    const float* bq      = (const float*)d_in[7];
    const float* Wk      = (const float*)d_in[8];
    const float* bk      = (const float*)d_in[9];
    const float* Wv      = (const float*)d_in[10];
    const float* bv      = (const float*)d_in[11];
    const float* Wp      = (const float*)d_in[12];
    const float* bp      = (const float*)d_in[13];
    const float* cbias   = (const float*)d_in[14];
    const float* pbias   = (const float*)d_in[15];
    const float* Wo      = (const float*)d_in[16];
    const float* bo      = (const float*)d_in[17];
    const float* Wf1     = (const float*)d_in[18];
    const float* bf1     = (const float*)d_in[19];
    const float* Wf2     = (const float*)d_in[20];
    const float* bf2     = (const float*)d_in[21];
    const float* ln_w    = (const float*)d_in[22];
    const float* ln_b    = (const float*)d_in[23];
    float* outp = (float*)d_out;

    // ---- workspace layout ----
    float* ws = (float*)d_ws;
    const size_t ACT = (size_t)B_ * L_ * HID_;            // 2M f32
    float* h  = ws;
    float* t1 = ws + ACT;
    float* q  = ws + 2 * ACT;       // reused for Wo / Wf2 outputs
    float* kb = ws + 3 * ACT;
    float* vb = ws + 4 * ACT;
    float* vT = ws + 5 * ACT;
    float* Pm = ws + 6 * ACT;                              // [L, HID]
    float* at = Pm + (size_t)L_ * HID_;
    float* h1 = at + ACT;
    // f16 transposed-weight region after 17M floats
    _Float16* w16 = (_Float16*)(h1 + ACT);
    const size_t W1M = (size_t)HID_ * HID_;                // 1M
    _Float16* tWh1 = w16;                                  // [1024][512]
    _Float16* tWh2 = tWh1 + (size_t)HID_ * IN_;            // [1024][1024]
    _Float16* tLyr = tWh2 + W1M;                           // 7 x 1M per layer

    const int ML = B_ * L_;

    // ---- repack all weights to f16 [N][K] (every call; deterministic) ----
    repack_w<<<(int)(((size_t)HID_ * IN_) / 256), 256, 0, stream>>>(Wh1, tWh1, IN_, HID_, 0);
    repack_w<<<(int)(W1M / 256), 256, 0, stream>>>(Wh2, tWh2, HID_, HID_, 0);
    for (int l = 0; l < NL_; ++l) {
        _Float16* base = tLyr + (size_t)l * 7 * W1M;
        repack_w<<<(int)(W1M / 256), 256, 0, stream>>>(Wq + (size_t)l * W1M, base + 0 * W1M, HID_, HID_, 1);
        repack_w<<<(int)(W1M / 256), 256, 0, stream>>>(Wk + (size_t)l * W1M, base + 1 * W1M, HID_, HID_, 1);
        repack_w<<<(int)(W1M / 256), 256, 0, stream>>>(Wv + (size_t)l * W1M, base + 2 * W1M, HID_, HID_, 1);
        repack_w<<<(int)(W1M / 256), 256, 0, stream>>>(Wp + (size_t)l * W1M, base + 3 * W1M, HID_, HID_, 1);
        repack_w<<<(int)(W1M / 256), 256, 0, stream>>>(Wo + (size_t)l * W1M, base + 4 * W1M, HID_, HID_, 0);
        repack_w<<<(int)(W1M / 256), 256, 0, stream>>>(Wf1 + (size_t)l * W1M, base + 5 * W1M, HID_, HID_, 0);
        repack_w<<<(int)(W1M / 256), 256, 0, stream>>>(Wf2 + (size_t)l * W1M, base + 6 * W1M, HID_, HID_, 0);
    }

    // grid helper: waves = (M/16)*(N/64), 4 waves per block
    const int GMM = (ML / 16) * (HID_ / 64) / 4;          // 2048x1024 GEMMs -> 512 blocks
    const int GPP = (L_ / 16) * (HID_ / 64) / 4;          // 1024x1024 GEMM  -> 256 blocks

    // to_hidden MLP
    gemm_nt<1><<<GMM, 128, 0, stream>>>(x,  IN_,  tWh1, IN_,  bh1, t1, HID_, ML, HID_, IN_);
    gemm_nt<0><<<GMM, 128, 0, stream>>>(t1, HID_, tWh2, HID_, bh2, h,  HID_, ML, HID_, HID_);

    for (int l = 0; l < NL_; ++l) {
        _Float16* base = tLyr + (size_t)l * 7 * W1M;
        const float* bq_l = bq + (size_t)l * H_ * DH_;
        const float* bk_l = bk + (size_t)l * H_ * DH_;
        const float* bv_l = bv + (size_t)l * H_ * DH_;
        const float* bp_l = bp + (size_t)l * H_ * DH_;
        const float* cb_l = cbias + (size_t)l * H_ * DH_;
        const float* pb_l = pbias + (size_t)l * H_ * DH_;
        const float* bo_l = bo + (size_t)l * HID_;
        const float* bf1_l = bf1 + (size_t)l * HID_;
        const float* bf2_l = bf2 + (size_t)l * HID_;
        const float* lw = ln_w + (size_t)l * HID_;
        const float* lb = ln_b + (size_t)l * HID_;

        // projections
        gemm_nt<0><<<GMM, 128, 0, stream>>>(h, HID_, base + 0 * W1M, HID_, bq_l, q,  HID_, ML, HID_, HID_);
        gemm_nt<0><<<GMM, 128, 0, stream>>>(h, HID_, base + 1 * W1M, HID_, bk_l, kb, HID_, ML, HID_, HID_);
        gemm_nt<0><<<GMM, 128, 0, stream>>>(h, HID_, base + 2 * W1M, HID_, bv_l, vb, HID_, ML, HID_, HID_);
        gemm_nt<0><<<GPP, 128, 0, stream>>>(pos_raw, HID_, base + 3 * W1M, HID_, bp_l, Pm, HID_, L_, HID_, HID_);

        transpose_v_kernel<<<(B_ * H_ * DH_ * L_) / 256, 256, 0, stream>>>(vb, vT);

        attn_kernel<<<B_ * H_ * (L_ / 16), 32, 0, stream>>>(q, kb, vT, Pm, cb_l, pb_l, at);

        gemm_nt<0><<<GMM, 128, 0, stream>>>(at, HID_, base + 4 * W1M, HID_, bo_l, q, HID_, ML, HID_, HID_);
        resid_ln_kernel<<<ML, 256, 0, stream>>>(q, h, lw, lb, h1);

        gemm_nt<1><<<GMM, 128, 0, stream>>>(h1, HID_, base + 5 * W1M, HID_, bf1_l, t1, HID_, ML, HID_, HID_);
        gemm_nt<0><<<GMM, 128, 0, stream>>>(t1, HID_, base + 6 * W1M, HID_, bf2_l, q,  HID_, ML, HID_, HID_);
        resid_ln_kernel<<<ML, 256, 0, stream>>>(h1, q, lw, lb, (l == NL_ - 1) ? outp : h);
    }
}
